// GAT_39934605919018
// MI455X (gfx1250) — matrix operand order repaired and checked by reading, not compiled, not run on hardware
//
#include <hip/hip_runtime.h>
#include <hip/hip_bf16.h>
#include <stdint.h>
#include <stddef.h>

// ---------------------------------------------------------------------------
// Problem constants (match reference)
// ---------------------------------------------------------------------------
namespace {
constexpr int kN   = 100000;   // nodes
constexpr int kE   = 500000;   // edges
constexpr int kDin = 64;       // input feature dim
constexpr int kH0  = 4;        // layer0 heads
constexpr int kF   = 64;       // feats per head
constexpr int kC0  = kH0 * kF; // 256 (layer0 concat width)
constexpr int kC1  = 2 * kF;   // 128 = [proj1(64) | skip1(64)]
}

typedef _Float16 v16h_t __attribute__((ext_vector_type(16)));
typedef float    v8f_t  __attribute__((ext_vector_type(8)));
typedef _Float16 h2_t   __attribute__((ext_vector_type(2)));
typedef unsigned int u32x4_t __attribute__((ext_vector_type(4)));
typedef int          i32x4_t __attribute__((ext_vector_type(4)));
typedef int          i32x8_t __attribute__((ext_vector_type(8)));

// Order-preserving float<->int encoding for atomic max on signed ints.
__device__ __forceinline__ int enc_ord(float f) {
  int i = __float_as_int(f);
  return i >= 0 ? i : (i ^ 0x7FFFFFFF);
}
__device__ __forceinline__ float dec_ord(int i) {
  return __int_as_float(i >= 0 ? i : (i ^ 0x7FFFFFFF));
}

// ---------------------------------------------------------------------------
// Utility kernels
// ---------------------------------------------------------------------------
__global__ void k_fill_zero(float* __restrict__ p, int n) {
  for (int i = blockIdx.x * blockDim.x + threadIdx.x; i < n;
       i += gridDim.x * blockDim.x)
    p[i] = 0.0f;
}

__global__ void k_init_gmax(int* __restrict__ g) {
  g[0] = (int)0x80000000;
  g[1] = (int)0x80000000;
}

__global__ void k_cvt_f16(const float* __restrict__ in,
                          _Float16* __restrict__ out, int n) {
  for (int i = blockIdx.x * blockDim.x + threadIdx.x; i < n;
       i += gridDim.x * blockDim.x)
    out[i] = (_Float16)in[i];
}

// ---------------------------------------------------------------------------
// TDM: DMA a [rows x K] f16 tile (row-major, stride K) from global into LDS.
// D# layout per CDNA5 ISA ch.8:
//   group0: [1:0]=count=1, [63:32]=lds_addr, [120:64]=global_addr, [127:126]=2
//   group1: [17:16]=data_size(1=2B), [79:48]=tensor_dim0, [111:80]=tensor_dim1,
//           [127:112]=tile_dim0, [143:128]=tile_dim1, [207:160]=dim0_stride
//   groups 2/3: zero (2-D tensor)
// ---------------------------------------------------------------------------
#if __has_builtin(__builtin_amdgcn_tensor_load_to_lds)
#define GAT_HAS_TDM 1
#else
#define GAT_HAS_TDM 0
#endif

__device__ __forceinline__ void tdm_load_tile(const _Float16* gsrc,
                                              unsigned int ldsOff, int K,
                                              int rows) {
#if GAT_HAS_TDM
  unsigned long long ga = (unsigned long long)(uintptr_t)gsrc;
  u32x4_t g0;
  g0[0] = 1u;  // count=1 (valid), user mode, gather off
  g0[1] = ldsOff;
  g0[2] = (unsigned int)(ga & 0xFFFFFFFFull);
  g0[3] = (unsigned int)((ga >> 32) & 0x01FFFFFFull) | 0x80000000u;  // type=2
  i32x8_t g1 = {};
  g1[0] = (1 << 16);                          // data_size = 2 bytes
  g1[1] = (int)((K & 0xFFFF) << 16);          // tensor_dim0 [15:0]
  g1[2] = (int)(((unsigned)K >> 16) & 0xFFFF) |
          (int)((rows & 0xFFFF) << 16);       // dim0 hi | tensor_dim1 lo
  g1[3] = (int)((K & 0xFFFF) << 16);          // tensor_dim1 hi=0 | tile_dim0
  g1[4] = (int)(rows & 0xFFFF);               // tile_dim1 (tile_dim2 = 0)
  g1[5] = K;                                  // tensor_dim0_stride [31:0]
  g1[6] = 0;
  g1[7] = 0;
  i32x4_t z4 = {};
#if __clang_major__ >= 23
  i32x8_t z8 = {};
  __builtin_amdgcn_tensor_load_to_lds(g0, g1, z4, z4, z8, 0);
#else
  __builtin_amdgcn_tensor_load_to_lds(g0, g1, z4, z4, 0);
#endif
  __builtin_amdgcn_s_wait_tensorcnt(0);
  asm volatile("" ::: "memory");
#else
  (void)gsrc; (void)ldsOff; (void)K; (void)rows;
#endif
}

// ---------------------------------------------------------------------------
// WMMA GEMM: C[M,NC] (f32) = A[M,K] (f16, row-major) x W[NC,K]^T (f16)
// One wave (32 threads) per 16-row strip:
//   1) TDM-load the 16xK A tile into LDS (tensor_load_to_lds + s_wait_tensorcnt)
//   2) build all K/32 A fragments once into registers (ds_load)
//   3) sweep NC/16 column tiles; per tile stream B fragments from the tiny
//      L2-resident weight matrix and issue v_wmma_f32_16x16x32_f16.
// Fragment layouts per CDNA5 ISA 7.12.2:
//   A 16x32 f16 : lane (m=lane&15, hi=lane>>4); vgpr v holds
//                 K = (v>>2)*16 + hi*8 + (v&3)*2  (+0,+1)
//   B 32x16 f16 : lane = column (lane&15); same K order as A:
//                 vgpr v holds K = (v>>2)*16 + hi*8 + (v&3)*2  (+0,+1)
//   D 16x16 f32 : vgpr v -> row M = hi*8 + v, col N = lane&15
// ---------------------------------------------------------------------------
template <int K>
__global__ __launch_bounds__(32)
void k_wmma_gemm(const _Float16* __restrict__ A, const _Float16* __restrict__ W,
                 float* __restrict__ C, int NC) {
  __shared__ _Float16 ldsA[16 * K];
  const int lane = threadIdx.x;
  const int m  = lane & 15;
  const int hi = lane >> 4;
  const int row0 = blockIdx.x << 4;

#if GAT_HAS_TDM
  tdm_load_tile(A + (size_t)row0 * K, (unsigned int)(uintptr_t)ldsA, K, 16);
#else
  {  // fallback: cooperative global->LDS copy (dword granularity)
    const unsigned int* src = (const unsigned int*)(A + (size_t)row0 * K);
    unsigned int* dst = (unsigned int*)ldsA;
    for (int i = lane; i < 16 * K / 2; i += 32) dst[i] = src[i];
  }
#endif

  constexpr int KSTEPS = K / 32;
  v16h_t af[KSTEPS];
#pragma unroll
  for (int s = 0; s < KSTEPS; ++s) {
    const _Float16* base = ldsA + m * K + s * 32;
#pragma unroll
    for (int v = 0; v < 8; ++v) {
      const int ka = ((v >> 2) << 4) + (hi << 3) + ((v & 3) << 1);
      h2_t p = *(const h2_t*)(base + ka);
      af[s][2 * v]     = p[0];
      af[s][2 * v + 1] = p[1];
    }
  }

  for (int col0 = 0; col0 < NC; col0 += 16) {
    const _Float16* __restrict__ wrow =
        W + (size_t)(col0 + m) * K + (hi << 3);
    v8f_t acc = {};
#pragma unroll
    for (int s = 0; s < KSTEPS; ++s) {
      v16h_t bf;
#pragma unroll
      for (int v = 0; v < 8; ++v) {
        h2_t p = *(const h2_t*)(wrow + s * 32 + ((v >> 2) << 4) + ((v & 3) << 1));
        bf[2 * v]     = p[0];
        bf[2 * v + 1] = p[1];
      }
      // (neg_a, A, neg_b, B, c_mod, C, reuse_a, reuse_b)
      acc = __builtin_amdgcn_wmma_f32_16x16x32_f16(false, af[s], false, bf,
                                                   (short)0, acc, false, false);
    }
#pragma unroll
    for (int v = 0; v < 8; ++v)
      C[(size_t)(row0 + (hi << 3) + v) * NC + col0 + m] = acc[v];
  }
}

// ---------------------------------------------------------------------------
// Per-node attention scores: s{1,2,3}[n,h] = dot(proj[n, h*F : h*F+F], a{1,2,3}[h])
// ---------------------------------------------------------------------------
__global__ void k_node_scores(const float* __restrict__ proj,
                              const float* __restrict__ a1,
                              const float* __restrict__ a2,
                              const float* __restrict__ a3,
                              float* __restrict__ s1, float* __restrict__ s2,
                              float* __restrict__ s3, int n, int H,
                              int rowStride) {
  int i = blockIdx.x * blockDim.x + threadIdx.x;  // i = node*H + h
  if (i >= n * H) return;
  int node = i / H, h = i - node * H;
  const float* __restrict__ p  = proj + (size_t)node * rowStride + h * kF;
  const float* __restrict__ w1 = a1 + h * kF;
  const float* __restrict__ w2 = a2 + h * kF;
  const float* __restrict__ w3 = a3 + h * kF;
  float d1 = 0.f, d2 = 0.f, d3 = 0.f;
#pragma unroll 8
  for (int f = 0; f < kF; ++f) {
    float v = p[f];
    d1 += v * w1[f];
    d2 += v * w2[f];
    d3 += v * w3[f];
  }
  s1[i] = d1; s2[i] = d2; s3[i] = d3;
}

// ---------------------------------------------------------------------------
// Global max of leaky_relu(s1[drug]+s2[tgt]+s3[dis]) over all (edge, head).
// ---------------------------------------------------------------------------
__global__ void k_edge_max(const int* __restrict__ ei,
                           const float* __restrict__ s1,
                           const float* __restrict__ s2,
                           const float* __restrict__ s3, int E, int H,
                           int* __restrict__ gmax) {
  __shared__ int smax;
  if (threadIdx.x == 0) smax = (int)0x80000000;
  __syncthreads();
  int e = blockIdx.x * blockDim.x + threadIdx.x;
  int loc = (int)0x80000000;
  if (e < E) {
    int d = ei[e], t = ei[E + e], z = ei[2 * E + e];
    for (int h = 0; h < H; ++h) {
      float sc = s1[(size_t)d * H + h] + s2[(size_t)t * H + h] +
                 s3[(size_t)z * H + h];
      sc = sc > 0.f ? sc : 0.2f * sc;
      int ev = enc_ord(sc);
      loc = loc > ev ? loc : ev;
    }
  }
  atomicMax(&smax, loc);
  __syncthreads();
  if (threadIdx.x == 0) atomicMax(gmax, smax);
}

// ---------------------------------------------------------------------------
// e = exp(score - gmax); scatter denominator into sums[tgt], sums[dis].
// ---------------------------------------------------------------------------
__global__ void k_edge_expsum(const int* __restrict__ ei,
                              const float* __restrict__ s1,
                              const float* __restrict__ s2,
                              const float* __restrict__ s3,
                              const int* __restrict__ gmax,
                              float* __restrict__ eout,
                              float* __restrict__ sums, int E, int H) {
  int e = blockIdx.x * blockDim.x + threadIdx.x;
  if (e >= E) return;
  float gm = dec_ord(*gmax);
  int d = ei[e], t = ei[E + e], z = ei[2 * E + e];
  for (int h = 0; h < H; ++h) {
    float sc = s1[(size_t)d * H + h] + s2[(size_t)t * H + h] +
               s3[(size_t)z * H + h];
    sc = sc > 0.f ? sc : 0.2f * sc;
    float ex = __expf(sc - gm);
    eout[(size_t)e * H + h] = ex;
    atomicAdd(&sums[(size_t)t * H + h], ex);
    atomicAdd(&sums[(size_t)z * H + h], ex);
  }
}

// ---------------------------------------------------------------------------
// Message scatter: 64 consecutive threads handle one edge (coalesced gather of
// proj[drug] rows; float atomics land in L2-resident accumulator).
// ---------------------------------------------------------------------------
__global__ void k_edge_msg(const int* __restrict__ ei,
                           const float* __restrict__ proj,
                           const float* __restrict__ eArr,
                           const float* __restrict__ sums,
                           float* __restrict__ acc, int E, int H, int strideP,
                           int strideA) {
  int idx = blockIdx.x * blockDim.x + threadIdx.x;
  int f = idx & (kF - 1);
  int e = idx >> 6;
  if (e >= E) return;
  int d = ei[e], t = ei[E + e], z = ei[2 * E + e];
  const float* __restrict__ prow = proj + (size_t)d * strideP;
  for (int h = 0; h < H; ++h) {
    float att = eArr[(size_t)e * H + h] /
                (sums[(size_t)t * H + h] + sums[(size_t)z * H + h] + 1e-16f);
    float msg = prow[h * kF + f] * att;
    atomicAdd(&acc[(size_t)t * strideA + h * kF + f], msg);
    atomicAdd(&acc[(size_t)z * strideA + h * kF + f], msg);
  }
}

// ---------------------------------------------------------------------------
// Layer-0 epilogue: h = elu(acc0 + x (identity skip, broadcast over heads) + b0)
// emitted directly as f16 for the next WMMA GEMM.
// ---------------------------------------------------------------------------
__global__ void k_epilogue0(const float* __restrict__ acc0,
                            const float* __restrict__ x,
                            const float* __restrict__ b0,
                            _Float16* __restrict__ hh, int total) {
  int i = blockIdx.x * blockDim.x + threadIdx.x;
  if (i >= total) return;
  int n = i >> 8;  // / 256
  int c = i & 255;
  float v = acc0[i] + x[(size_t)n * kDin + (c & (kF - 1))] + b0[c];
  v = v > 0.f ? v : (__expf(v) - 1.0f);  // ELU (alpha = 1)
  hh[i] = (_Float16)v;
}

// ---------------------------------------------------------------------------
// Layer-1 epilogue: out = acc1 + skip1 + b1  (mean over 1 head == identity)
// ps1 rows: [ proj1 (64) | skip1 (64) ], stride 128.
// ---------------------------------------------------------------------------
__global__ void k_final(const float* __restrict__ acc1,
                        const float* __restrict__ ps1,
                        const float* __restrict__ b1, float* __restrict__ out,
                        int total) {
  int i = blockIdx.x * blockDim.x + threadIdx.x;
  if (i >= total) return;
  int n = i >> 6;
  int f = i & (kF - 1);
  out[i] = acc1[i] + ps1[(size_t)n * kC1 + kF + f] + b1[f];
}

// ---------------------------------------------------------------------------
// Host-side orchestration
// ---------------------------------------------------------------------------
static inline int cdiv(int a, int b) { return (a + b - 1) / b; }

extern "C" void kernel_launch(void* const* d_in, const int* in_sizes, int n_in,
                              void* d_out, int out_size, void* d_ws,
                              size_t ws_size, hipStream_t stream) {
  (void)in_sizes; (void)n_in; (void)out_size; (void)ws_size;

  const float* x      = (const float*)d_in[0];   // [N,64]
  const int*   ei     = (const int*)d_in[1];     // [3,E]
  const float* W0     = (const float*)d_in[2];   // [256,64]
  const float* a10    = (const float*)d_in[3];   // [4,64]
  const float* a20    = (const float*)d_in[4];
  const float* a30    = (const float*)d_in[5];
  // d_in[6] = Wskip0 is dead in the reference (layer0 uses identity skip)
  const float* b0     = (const float*)d_in[7];   // [256]
  const float* W1     = (const float*)d_in[8];   // [64,256]
  const float* a11    = (const float*)d_in[9];   // [1,64]
  const float* a21    = (const float*)d_in[10];
  const float* a31    = (const float*)d_in[11];
  const float* Wskip1 = (const float*)d_in[12];  // [64,256]
  const float* b1     = (const float*)d_in[13];  // [64]
  float* out = (float*)d_out;                    // [N,64]

  // ---- workspace carve-out (256B aligned slabs) ----
  char*  ws  = (char*)d_ws;
  size_t off = 0;
  auto alloc = [&](size_t bytes) -> void* {
    void* p = ws + off;
    off = (off + bytes + 255) & ~(size_t)255;
    return p;
  };
  _Float16* xh  = (_Float16*)alloc((size_t)kN * kDin * 2);   // 12.8 MB
  _Float16* w0h = (_Float16*)alloc((size_t)kC0 * kDin * 2);  // 32 KB
  _Float16* w1h = (_Float16*)alloc((size_t)kC1 * kC0 * 2);   // 64 KB
  float* proj0  = (float*)alloc((size_t)kN * kC0 * 4);       // 102.4 MB
  float* acc0   = (float*)alloc((size_t)kN * kC0 * 4);       // 102.4 MB
  float* s10    = (float*)alloc((size_t)kN * kH0 * 4);
  float* s20    = (float*)alloc((size_t)kN * kH0 * 4);
  float* s30    = (float*)alloc((size_t)kN * kH0 * 4);
  float* sums0  = (float*)alloc((size_t)kN * kH0 * 4);
  float* e0     = (float*)alloc((size_t)kE * kH0 * 4);       // 8 MB
  float* s11    = (float*)alloc((size_t)kN * 4);
  float* s21    = (float*)alloc((size_t)kN * 4);
  float* s31    = (float*)alloc((size_t)kN * 4);
  float* sums1  = (float*)alloc((size_t)kN * 4);
  float* e1     = (float*)alloc((size_t)kE * 4);
  float* acc1   = (float*)alloc((size_t)kN * kF * 4);        // 25.6 MB
  int*   gmax   = (int*)alloc(2 * sizeof(int));
  // Reuse (stream-ordered, no overlap of live ranges):
  _Float16* hh  = (_Float16*)proj0;  // layer-0 activations (f16) over proj0
  float*    ps1 = acc0;              // layer-1 [proj1|skip1] over acc0

  // ---- init (ws is poisoned; re-init every call for determinism) ----
  k_fill_zero<<<2048, 256, 0, stream>>>(acc0, kN * kC0);
  k_fill_zero<<<1024, 256, 0, stream>>>(acc1, kN * kF);
  k_fill_zero<<<512, 256, 0, stream>>>(sums0, kN * kH0);
  k_fill_zero<<<512, 256, 0, stream>>>(sums1, kN);
  k_init_gmax<<<1, 1, 0, stream>>>(gmax);

  // ---- f16 packing ----
  k_cvt_f16<<<2048, 256, 0, stream>>>(x, xh, kN * kDin);
  k_cvt_f16<<<64, 256, 0, stream>>>(W0, w0h, kC0 * kDin);
  k_cvt_f16<<<64, 256, 0, stream>>>(W1, w1h, kF * kC0);               // cols 0..63
  k_cvt_f16<<<64, 256, 0, stream>>>(Wskip1, w1h + (size_t)kF * kC0,   // cols 64..127
                                    kF * kC0);

  // ---- layer 0 ----
  k_wmma_gemm<kDin><<<kN / 16, 32, 0, stream>>>(xh, w0h, proj0, kC0);
  k_node_scores<<<cdiv(kN * kH0, 256), 256, 0, stream>>>(
      proj0, a10, a20, a30, s10, s20, s30, kN, kH0, kC0);
  k_edge_max<<<cdiv(kE, 256), 256, 0, stream>>>(ei, s10, s20, s30, kE, kH0,
                                                gmax + 0);
  k_edge_expsum<<<cdiv(kE, 256), 256, 0, stream>>>(ei, s10, s20, s30, gmax + 0,
                                                   e0, sums0, kE, kH0);
  k_edge_msg<<<cdiv(kE * kF, 256), 256, 0, stream>>>(ei, proj0, e0, sums0,
                                                     acc0, kE, kH0, kC0, kC0);
  k_epilogue0<<<cdiv(kN * kC0, 256), 256, 0, stream>>>(acc0, x, b0, hh,
                                                       kN * kC0);

  // ---- layer 1 ----
  k_wmma_gemm<kC0><<<kN / 16, 32, 0, stream>>>(hh, w1h, ps1, kC1);
  k_node_scores<<<cdiv(kN, 256), 256, 0, stream>>>(ps1, a11, a21, a31, s11,
                                                   s21, s31, kN, 1, kC1);
  k_edge_max<<<cdiv(kE, 256), 256, 0, stream>>>(ei, s11, s21, s31, kE, 1,
                                                gmax + 1);
  k_edge_expsum<<<cdiv(kE, 256), 256, 0, stream>>>(ei, s11, s21, s31, gmax + 1,
                                                   e1, sums1, kE, 1);
  k_edge_msg<<<cdiv(kE * kF, 256), 256, 0, stream>>>(ei, ps1, e1, sums1, acc1,
                                                     kE, 1, kC1, kF);
  k_final<<<cdiv(kN * kF, 256), 256, 0, stream>>>(acc1, ps1, b1, out,
                                                  kN * kF);
}
